// MambaActor_6176162972365
// MI455X (gfx1250) — matrix operand order repaired and checked by reading, not compiled, not run on hardware
//
#include <hip/hip_runtime.h>
#include <hip/hip_bf16.h>

// ---------------------------------------------------------------------------
// Mamba-2 forward for MI455X (gfx1250, wave32, WMMA bf16 16x16x32, f32 accum)
// TDM (tensor_load_to_lds) stages SSD tiles; GEMMs are register-double-buffered.
// ---------------------------------------------------------------------------
// B=128 L=512 CIN=48 COUT=12 D_MODEL=256 N_LAYER=4 D_STATE=64 D_CONV=4
// D_INNER=512 NHEADS=8 HEADDIM=64 CHUNK=64 CONV_DIM=640 D_IN_PROJ=1160

typedef __attribute__((ext_vector_type(16))) __bf16 v16bf;
typedef __attribute__((ext_vector_type(8)))  __bf16 v8bf;
typedef __attribute__((ext_vector_type(8)))  float  v8f;
typedef unsigned __attribute__((ext_vector_type(4))) uint4v;
typedef int      __attribute__((ext_vector_type(8))) int8v;
typedef int      __attribute__((ext_vector_type(4))) int4v;

#define NROWS   65536
#define DMODEL  256
#define DINNER  512
#define NPROJ   1160
#define NPROJP  1280   // padded to 16-col tiles for WMMA
#define NHEADS  8
#define HD      64
#define NSTATE  64
#define CHUNK   64
#define NCHUNK  8
#define NBATCH  128

__device__ __forceinline__ __bf16 f2bf(float f) {
  unsigned u = __builtin_bit_cast(unsigned, f);
  unsigned r = u + 0x7FFFu + ((u >> 16) & 1u);      // round-to-nearest-even
  unsigned short h = (unsigned short)(r >> 16);
  return __builtin_bit_cast(__bf16, h);
}
__device__ __forceinline__ float bf2f(__bf16 b) {
  unsigned short s = __builtin_bit_cast(unsigned short, b);
  unsigned u = ((unsigned)s) << 16;
  return __builtin_bit_cast(float, u);
}
__device__ __forceinline__ v8f vzero8() {
  v8f z = {0.f,0.f,0.f,0.f,0.f,0.f,0.f,0.f};
  return z;
}
__device__ __forceinline__ v8f wmma_bf16(v16bf a, v16bf b, v8f c) {
  return __builtin_amdgcn_wmma_f32_16x16x32_bf16(false, a, false, b, (short)0, c,
                                                 false, false);
}

// A fragment: 16(M)x32(K) bf16, source row-major, contiguous K, leading dim ld.
__device__ __forceinline__ v16bf load_frag_A(const __bf16* p, int ld, int lane) {
  int row = lane & 15;
  int kb  = (lane >> 4) << 3;
  const __bf16* q = p + (size_t)row * ld + kb;
  v8bf lo = *(const v8bf*)(q);
  v8bf hi = *(const v8bf*)(q + 16);
  v16bf f;
#pragma unroll
  for (int i = 0; i < 8; i++) { f[i] = lo[i]; f[i + 8] = hi[i]; }
  return f;
}

// B fragment: 32(K)x16(N) bf16, source is B^T row-major (N x K), contiguous K.
__device__ __forceinline__ v16bf load_frag_B(const __bf16* p, int ld, int lane) {
  int col = lane & 15;
  int kb  = (lane >> 4) << 3;
  const __bf16* q = p + (size_t)col * ld + kb;
  v8bf lo = *(const v8bf*)(q);
  v8bf hi = *(const v8bf*)(q + 16);
  v16bf f;
#pragma unroll
  for (int i = 0; i < 8; i++) { f[i] = lo[i]; f[i + 8] = hi[i]; }
  return f;
}

__device__ __forceinline__ float wave_sum(float v) {
#pragma unroll
  for (int o = 16; o > 0; o >>= 1) v += __shfl_xor(v, o, 32);
  return v;
}

// ---------------------------------------------------------------------------
// Tensor Data Mover: 2D tile (tile_d1 rows x tile_d0 bf16 elems, row stride
// stride_elems) from global -> LDS at byte offset lds_off. D# per ISA 8.3-8.5.
// clang-23 lane: 6-arg builtin (g0, g1, g2, g3, g4, cpol).
// ---------------------------------------------------------------------------
__device__ __forceinline__ void tdm_load_2d_bf16(unsigned lds_off, const void* gptr,
                                                 unsigned tile_d0, unsigned tile_d1,
                                                 unsigned stride_elems) {
  unsigned long long ga = (unsigned long long)(uintptr_t)gptr;
  uint4v g0;
  g0[0] = 1u;                                   // count=1, user descriptor
  g0[1] = lds_off;                              // lds_addr (bytes)
  g0[2] = (unsigned)(ga & 0xFFFFFFFFu);         // global_addr[31:0]
  g0[3] = (unsigned)((ga >> 32) & 0x01FFFFFFu)  // global_addr[56:32]
        | (2u << 30);                           // type=2 (image)
  int8v g1;
  g1[0] = 0x00010000;                           // data_size=1 (2 bytes)
  g1[1] = (int)(tile_d0 << 16);                 // tensor_dim0[15:0] in [31:16]
  g1[2] = (int)(tile_d1 << 16);                 // tensor_dim1[15:0] in [31:16]
  g1[3] = (int)(tile_d0 << 16);                 // tile_dim0 in [31:16]
  g1[4] = (int)tile_d1;                         // tile_dim1 (tile_dim2=0)
  g1[5] = (int)stride_elems;                    // tensor_dim0_stride[31:0]
  g1[6] = 0;
  g1[7] = 0;
  int4v z4 = {0, 0, 0, 0};
  int8v z8 = {0, 0, 0, 0, 0, 0, 0, 0};
  __builtin_amdgcn_tensor_load_to_lds(g0, g1, z4, z4, z8, 0);
}

// ---------------------------------------------------------------------------
// Weight conversion: Wi (4,1160,256) f32 -> padded bf16 (4,1280,256);
// Wo (4,256,512) f32 -> bf16.
// ---------------------------------------------------------------------------
__global__ void k_wcvt(const float* __restrict__ Wi, const float* __restrict__ Wo,
                       __bf16* __restrict__ WiB, __bf16* __restrict__ WoB) {
  size_t i = (size_t)blockIdx.x * blockDim.x + threadIdx.x;
  const size_t nWiB = (size_t)4 * NPROJP * DMODEL;
  const size_t nWoB = (size_t)4 * DMODEL * DINNER;
  if (i < nWiB) {
    int col = (int)(i % DMODEL);
    size_t r = i / DMODEL;
    int n  = (int)(r % NPROJP);
    int ly = (int)(r / NPROJP);
    float v = (n < NPROJ) ? Wi[((size_t)ly * NPROJ + n) * DMODEL + col] : 0.f;
    WiB[i] = f2bf(v);
  }
  if (i < nWoB) WoB[i] = f2bf(Wo[i]);
}

// ---------------------------------------------------------------------------
// Embedding GEMM (K=48) + LayerNorm -> residual X (f32); one row per block
// ---------------------------------------------------------------------------
__global__ void k_embed_ln(const float* __restrict__ obs, const float* __restrict__ inW,
                           const float* __restrict__ lw, const float* __restrict__ lb,
                           float* __restrict__ X) {
  __shared__ float sObs[48];
  __shared__ float ws1[8], ws2[8];
  size_t row = blockIdx.x;
  int t = threadIdx.x;
  if (t < 48) sObs[t] = obs[row * 48 + t];
  __syncthreads();
  float acc = 0.f;
#pragma unroll
  for (int k = 0; k < 48; k++) acc += sObs[k] * inW[t * 48 + k];
  float s = wave_sum(acc);
  float s2 = wave_sum(acc * acc);
  int wv = t >> 5, lane = t & 31;
  if (lane == 0) { ws1[wv] = s; ws2[wv] = s2; }
  __syncthreads();
  float ts = 0.f, ts2 = 0.f;
#pragma unroll
  for (int i = 0; i < 8; i++) { ts += ws1[i]; ts2 += ws2[i]; }
  float mu  = ts  * (1.f / DMODEL);
  float var = ts2 * (1.f / DMODEL) - mu * mu;
  X[row * DMODEL + t] = (acc - mu) * rsqrtf(var + 1e-5f) * lw[t] + lb[t];
}

// ---------------------------------------------------------------------------
// RMSNorm X(f32) -> H(bf16), one row per wave, 8 rows per block
// ---------------------------------------------------------------------------
__global__ void k_rms(const float* __restrict__ X, const float* __restrict__ w,
                      __bf16* __restrict__ H) {
  int lane = threadIdx.x & 31, wv = threadIdx.x >> 5;
  size_t row = (size_t)blockIdx.x * 8 + wv;
  const float* xr = X + row * DMODEL;
  float v[8]; float ss = 0.f;
#pragma unroll
  for (int i = 0; i < 8; i++) { v[i] = xr[lane + i * 32]; ss += v[i] * v[i]; }
  ss = wave_sum(ss);
  float sc = rsqrtf(ss * (1.f / DMODEL) + 1e-5f);
#pragma unroll
  for (int i = 0; i < 8; i++)
    H[row * DMODEL + lane + i * 32] = f2bf(v[i] * sc * w[lane + i * 32]);
}

// ---------------------------------------------------------------------------
// In-projection GEMM: ZX[65536 x 1160] = H @ WiB^T (WiB: 1280 x 256 bf16)
// block = 8 waves -> 64(M) x 256(N); wave -> 32x64, register double-buffered
// ---------------------------------------------------------------------------
__global__ void k_gemm_in(const __bf16* __restrict__ A, const __bf16* __restrict__ W,
                          float* __restrict__ C) {
  int lane = threadIdx.x & 31, wv = threadIdx.x >> 5;
  size_t m0 = (size_t)blockIdx.x * 64 + (size_t)(wv & 1) * 32;
  int n0 = blockIdx.y * 256 + (wv >> 1) * 64;
  const __bf16* Ap = A + m0 * DMODEL;
  v8f acc[2][4];
#pragma unroll
  for (int i = 0; i < 2; i++)
#pragma unroll
    for (int j = 0; j < 4; j++) acc[i][j] = vzero8();

  v16bf a[2], b[4];
#pragma unroll
  for (int i = 0; i < 2; i++) a[i] = load_frag_A(Ap + (size_t)i * 16 * DMODEL, DMODEL, lane);
#pragma unroll
  for (int j = 0; j < 4; j++)
    b[j] = load_frag_B(W + (size_t)(n0 + j * 16) * DMODEL, DMODEL, lane);

#pragma unroll
  for (int kk = 0; kk < DMODEL; kk += 32) {
    v16bf an[2], bn[4];
    if (kk + 32 < DMODEL) {
#pragma unroll
      for (int i = 0; i < 2; i++)
        an[i] = load_frag_A(Ap + (size_t)i * 16 * DMODEL + kk + 32, DMODEL, lane);
#pragma unroll
      for (int j = 0; j < 4; j++)
        bn[j] = load_frag_B(W + (size_t)(n0 + j * 16) * DMODEL + kk + 32, DMODEL, lane);
    }
#pragma unroll
    for (int i = 0; i < 2; i++)
#pragma unroll
      for (int j = 0; j < 4; j++)
        acc[i][j] = wmma_bf16(a[i], b[j], acc[i][j]);
    if (kk + 32 < DMODEL) {
#pragma unroll
      for (int i = 0; i < 2; i++) a[i] = an[i];
#pragma unroll
      for (int j = 0; j < 4; j++) b[j] = bn[j];
    }
  }
#pragma unroll
  for (int i = 0; i < 2; i++)
#pragma unroll
    for (int j = 0; j < 4; j++) {
      int col = n0 + j * 16 + (lane & 15);
      if (col < NPROJ) {
        int rb = (lane >> 4) << 3;
#pragma unroll
        for (int r = 0; r < 8; r++)
          C[(m0 + i * 16 + rb + r) * NPROJ + col] = acc[i][j][r];
      }
    }
}

// ---------------------------------------------------------------------------
// Causal depthwise conv (D_CONV=4) + SiLU + softplus(dt)
// ---------------------------------------------------------------------------
__global__ void k_conv(const float* __restrict__ ZX, const float* __restrict__ cw,
                       const float* __restrict__ cb, const float* __restrict__ dtb,
                       const float* __restrict__ Alog,
                       __bf16* __restrict__ XDT, float* __restrict__ XS,
                       __bf16* __restrict__ Bb, __bf16* __restrict__ Cb,
                       float* __restrict__ ADT) {
  size_t row = blockIdx.x;
  int l = (int)(row & 511);
  int t = threadIdx.x;
  float acc = cb[t];
#pragma unroll
  for (int k = 0; k < 4; k++) {
    int lp = l + k - 3;
    if (lp >= 0) acc += ZX[(row - 3 + k) * NPROJ + DINNER + t] * cw[t * 4 + k];
  }
  float c = acc / (1.f + expf(-acc));          // SiLU
  if (t < DINNER) {
    int h = t >> 6;
    float dtr = ZX[row * NPROJ + (NPROJ - NHEADS) + h] + dtb[h];
    float dt = (dtr > 20.f) ? dtr : log1pf(expf(dtr));   // softplus
    XS[row * DINNER + t] = c;
    XDT[row * DINNER + t] = f2bf(c * dt);
    if ((t & 63) == 0) ADT[row * NHEADS + h] = -expf(Alog[h]) * dt;
  } else if (t < DINNER + NSTATE) {
    Bb[row * NSTATE + (t - DINNER)] = f2bf(c);
  } else {
    Cb[row * NSTATE + (t - DINNER - NSTATE)] = f2bf(c);
  }
}

// ---------------------------------------------------------------------------
// Per-chunk cumsum of A*dt: ACUM (b,h,c,64) and chunk sums CSUM (b,h,c)
// ---------------------------------------------------------------------------
__global__ void k_cumsum(const float* __restrict__ ADT, float* __restrict__ ACUM,
                         float* __restrict__ CSUM) {
  int idx = blockIdx.x;            // ((b*8+h)*8+c)
  int c = idx & 7, h = (idx >> 3) & 7, b = idx >> 6;
  __shared__ float sh[CHUNK];
  int l = threadIdx.x;
  sh[l] = ADT[((size_t)b * 512 + c * CHUNK + l) * NHEADS + h];
  __syncthreads();
  if (l == 0) {
    float s = 0.f;
    for (int i = 0; i < CHUNK; i++) { s += sh[i]; sh[i] = s; }
  }
  __syncthreads();
  ACUM[(size_t)idx * CHUNK + l] = sh[l];
  if (l == 0) CSUM[idx] = sh[CHUNK - 1];
}

// ---------------------------------------------------------------------------
// SSD intra-chunk kernel, one (b,c,h) per block, 128 threads (4 waves):
// TDM stages X and B tiles into LDS; then
//   G = C @ B^T (WMMA), M = G*exp(segsum) masked -> LDS bf16
//   Y_diag = M @ X (WMMA), states = X^T @ (B*decay) (WMMA)
// ---------------------------------------------------------------------------
__global__ void k_chunk(const __bf16* __restrict__ Cb, const __bf16* __restrict__ Bb,
                        const __bf16* __restrict__ XDT, const float* __restrict__ ACUM,
                        float* __restrict__ YD, float* __restrict__ SBUF) {
  const int LD = 72;                       // skewed LDS leading dim (16B aligned)
  __shared__ __bf16 sXraw[CHUNK * HD];     // TDM landing: (l, p) row-major
  __shared__ __bf16 sBraw[CHUNK * NSTATE]; // TDM landing: (l, n) row-major
  __shared__ __bf16 sBt[NSTATE * 72];      // (n, l) = B[l,n] * ds[l]
  __shared__ __bf16 sXt[HD * 72];          // (p, l) = X[l, h*64+p]
  __shared__ __bf16 sM[CHUNK * 72];        // (l, s)
  __shared__ float sAc[CHUNK];
  __shared__ float sDs[CHUNK];

  int bid = blockIdx.x;
  int h = bid & 7, c = (bid >> 3) & 7, b = bid >> 6;
  int t = threadIdx.x;
  int lane = t & 31, wv = t >> 5;
  size_t rowbase = (size_t)b * 512 + (size_t)c * CHUNK;
  size_t acbase = (size_t)(((b * 8 + h) * 8) + c) * CHUNK;

  // TDM: X tile (64 x 64 of stride-512 tensor) and B tile (64 x 64 contiguous)
  if (wv == 0) {
    tdm_load_2d_bf16((unsigned)(uintptr_t)&sXraw[0],
                     XDT + rowbase * DINNER + h * HD, HD, CHUNK, DINNER);
    tdm_load_2d_bf16((unsigned)(uintptr_t)&sBraw[0],
                     Bb + rowbase * NSTATE, NSTATE, CHUNK, NSTATE);
  }
  if (t < CHUNK) sAc[t] = ACUM[acbase + t];
  __syncthreads();
  if (t < CHUNK) sDs[t] = expf(sAc[CHUNK - 1] - sAc[t]);
  if (wv == 0) __builtin_amdgcn_s_wait_tensorcnt(0);
  __syncthreads();

  // transpose (+scale for B) from TDM landing buffers
  for (int i = t; i < CHUNK * NSTATE; i += 128) {
    int l = i >> 6, j = i & 63;
    float bv = bf2f(sBraw[l * NSTATE + j]) * sDs[l];
    sBt[j * LD + l] = f2bf(bv);
    sXt[j * LD + l] = sXraw[l * HD + j];
  }
  __syncthreads();

  // --- G = C @ B^T, then M = L * G ---
  int lt = wv * 16;
  v8f g[4];
#pragma unroll
  for (int i = 0; i < 4; i++) g[i] = vzero8();
  for (int kk = 0; kk < NSTATE; kk += 32) {
    v16bf a = load_frag_A(Cb + (rowbase + lt) * NSTATE + kk, NSTATE, lane);
#pragma unroll
    for (int s4 = 0; s4 < 4; s4++) {
      v16bf bb = load_frag_B(Bb + (rowbase + s4 * 16) * NSTATE + kk, NSTATE, lane);
      g[s4] = wmma_bf16(a, bb, g[s4]);
    }
  }
#pragma unroll
  for (int s4 = 0; s4 < 4; s4++) {
    int sc = s4 * 16 + (lane & 15);
    int rb = (lane >> 4) << 3;
#pragma unroll
    for (int r = 0; r < 8; r++) {
      int ll = lt + rb + r;
      float f = (ll >= sc) ? expf(sAc[ll] - sAc[sc]) : 0.f;
      sM[ll * LD + sc] = f2bf(g[s4][r] * f);
    }
  }
  __syncthreads();

  // --- Y_diag = M @ X ---
  v8f y[4];
#pragma unroll
  for (int i = 0; i < 4; i++) y[i] = vzero8();
  for (int kk = 0; kk < CHUNK; kk += 32) {
    v16bf a = load_frag_A(sM + lt * LD + kk, LD, lane);
#pragma unroll
    for (int p4 = 0; p4 < 4; p4++) {
      v16bf bb = load_frag_B(sXt + (p4 * 16) * LD + kk, LD, lane);
      y[p4] = wmma_bf16(a, bb, y[p4]);
    }
  }
#pragma unroll
  for (int p4 = 0; p4 < 4; p4++) {
    int p = p4 * 16 + (lane & 15);
    int rb = (lane >> 4) << 3;
#pragma unroll
    for (int r = 0; r < 8; r++)
      YD[(rowbase + lt + rb + r) * DINNER + h * HD + p] = y[p4][r];
  }

  // --- states[p,n] = sum_l X[l,p] * B[l,n] * ds[l] ---
  int pt = wv * 16;
  v8f s[4];
#pragma unroll
  for (int i = 0; i < 4; i++) s[i] = vzero8();
  for (int kk = 0; kk < CHUNK; kk += 32) {
    v16bf a = load_frag_A(sXt + pt * LD + kk, LD, lane);
#pragma unroll
    for (int n4 = 0; n4 < 4; n4++) {
      v16bf bb = load_frag_B(sBt + (n4 * 16) * LD + kk, LD, lane);
      s[n4] = wmma_bf16(a, bb, s[n4]);
    }
  }
  size_t sbase = (size_t)(((b * 8 + h) * 8) + c) * (HD * NSTATE);
#pragma unroll
  for (int n4 = 0; n4 < 4; n4++) {
    int n = n4 * 16 + (lane & 15);
    int rb = (lane >> 4) << 3;
#pragma unroll
    for (int r = 0; r < 8; r++)
      SBUF[sbase + (size_t)(pt + rb + r) * NSTATE + n] = s[n4][r];
  }
}

// ---------------------------------------------------------------------------
// Cross-chunk scan: P_0 = 0; emit P_z (bf16); P_{z+1} = P_z*exp(sum_z) + S_z
// ---------------------------------------------------------------------------
__global__ void k_scan(const float* __restrict__ SBUF, const float* __restrict__ CSUM,
                       __bf16* __restrict__ PBUF) {
  int bh = blockIdx.x;
  int t = threadIdx.x;
  float P[16];
#pragma unroll
  for (int i = 0; i < 16; i++) P[i] = 0.f;
#pragma unroll 1
  for (int z = 0; z < NCHUNK; z++) {
    size_t base = ((size_t)bh * NCHUNK + z) * (HD * NSTATE);
    float lam = expf(CSUM[bh * NCHUNK + z]);
#pragma unroll
    for (int i = 0; i < 16; i++) {
      int e = i * 256 + t;
      PBUF[base + e] = f2bf(P[i]);
      P[i] = P[i] * lam + SBUF[base + e];
    }
  }
}

// ---------------------------------------------------------------------------
// Y_off = C @ P^T (WMMA), scaled by exp(A_cum[l]); fold in Y_diag and D*xs.
// ---------------------------------------------------------------------------
__global__ void k_yoff(const __bf16* __restrict__ Cb, const __bf16* __restrict__ PBUF,
                       const float* __restrict__ ACUM, const float* __restrict__ XS,
                       const float* __restrict__ Dp, float* __restrict__ YD) {
  __shared__ float sAc[CHUNK];
  int bid = blockIdx.x;
  int h = bid & 7, c = (bid >> 3) & 7, b = bid >> 6;
  int t = threadIdx.x;
  int lane = t & 31, wv = t >> 5;
  size_t rowbase = (size_t)b * 512 + (size_t)c * CHUNK;
  size_t acbase = (size_t)(((b * 8 + h) * 8) + c) * CHUNK;
  if (t < CHUNK) sAc[t] = ACUM[acbase + t];
  __syncthreads();

  int lt = wv * 16;
  size_t pbase = (size_t)(((b * 8 + h) * 8) + c) * (HD * NSTATE);
  v8f y[4];
#pragma unroll
  for (int i = 0; i < 4; i++) y[i] = vzero8();
  for (int kk = 0; kk < NSTATE; kk += 32) {
    v16bf a = load_frag_A(Cb + (rowbase + lt) * NSTATE + kk, NSTATE, lane);
#pragma unroll
    for (int p4 = 0; p4 < 4; p4++) {
      v16bf bb = load_frag_B(PBUF + pbase + (size_t)(p4 * 16) * NSTATE + kk, NSTATE, lane);
      y[p4] = wmma_bf16(a, bb, y[p4]);
    }
  }
  float dh = Dp[h];
#pragma unroll
  for (int p4 = 0; p4 < 4; p4++) {
    int p = p4 * 16 + (lane & 15);
    int rb = (lane >> 4) << 3;
#pragma unroll
    for (int r = 0; r < 8; r++) {
      int ll = lt + rb + r;
      size_t idx = (rowbase + ll) * DINNER + h * HD + p;
      float ea = expf(sAc[ll]);
      YD[idx] = YD[idx] + y[p4][r] * ea + dh * XS[idx];
    }
  }
}

// ---------------------------------------------------------------------------
// Gating + RMSNorm over D_INNER: YG = rmsnorm(YD * silu(z), gw)  (bf16)
// ---------------------------------------------------------------------------
__global__ void k_gate(const float* __restrict__ YD, const float* __restrict__ ZX,
                       const float* __restrict__ gw, __bf16* __restrict__ YG) {
  int lane = threadIdx.x & 31, wv = threadIdx.x >> 5;
  size_t row = (size_t)blockIdx.x * 8 + wv;
  float v[16]; float ss = 0.f;
#pragma unroll
  for (int i = 0; i < 16; i++) {
    int ci = lane + i * 32;
    float z = ZX[row * NPROJ + ci];
    float g = YD[row * DINNER + ci] * (z / (1.f + expf(-z)));
    v[i] = g; ss += g * g;
  }
  ss = wave_sum(ss);
  float sc = rsqrtf(ss * (1.f / DINNER) + 1e-5f);
#pragma unroll
  for (int i = 0; i < 16; i++) {
    int ci = lane + i * 32;
    YG[row * DINNER + ci] = f2bf(v[i] * sc * gw[ci]);
  }
}

// ---------------------------------------------------------------------------
// Out-projection GEMM + residual: X[65536x256] += YG[65536x512] @ WoB^T
// block = 8 waves -> 64(M) x 256(N); wave -> 32x64, register double-buffered
// ---------------------------------------------------------------------------
__global__ void k_gemm_out(const __bf16* __restrict__ A, const __bf16* __restrict__ W,
                           float* __restrict__ X) {
  int lane = threadIdx.x & 31, wv = threadIdx.x >> 5;
  size_t m0 = (size_t)blockIdx.x * 64 + (size_t)(wv & 1) * 32;
  int n0 = (wv >> 1) * 64;
  const __bf16* Ap = A + m0 * DINNER;
  v8f acc[2][4];
#pragma unroll
  for (int i = 0; i < 2; i++)
#pragma unroll
    for (int j = 0; j < 4; j++) acc[i][j] = vzero8();

  v16bf a[2], b[4];
#pragma unroll
  for (int i = 0; i < 2; i++) a[i] = load_frag_A(Ap + (size_t)i * 16 * DINNER, DINNER, lane);
#pragma unroll
  for (int j = 0; j < 4; j++)
    b[j] = load_frag_B(W + (size_t)(n0 + j * 16) * DINNER, DINNER, lane);

#pragma unroll
  for (int kk = 0; kk < DINNER; kk += 32) {
    v16bf an[2], bn[4];
    if (kk + 32 < DINNER) {
#pragma unroll
      for (int i = 0; i < 2; i++)
        an[i] = load_frag_A(Ap + (size_t)i * 16 * DINNER + kk + 32, DINNER, lane);
#pragma unroll
      for (int j = 0; j < 4; j++)
        bn[j] = load_frag_B(W + (size_t)(n0 + j * 16) * DINNER + kk + 32, DINNER, lane);
    }
#pragma unroll
    for (int i = 0; i < 2; i++)
#pragma unroll
      for (int j = 0; j < 4; j++)
        acc[i][j] = wmma_bf16(a[i], b[j], acc[i][j]);
    if (kk + 32 < DINNER) {
#pragma unroll
      for (int i = 0; i < 2; i++) a[i] = an[i];
#pragma unroll
      for (int j = 0; j < 4; j++) b[j] = bn[j];
    }
  }
#pragma unroll
  for (int i = 0; i < 2; i++)
#pragma unroll
    for (int j = 0; j < 4; j++) {
      int col = n0 + j * 16 + (lane & 15);
      int rb = (lane >> 4) << 3;
#pragma unroll
      for (int r = 0; r < 8; r++) {
        size_t idx = (m0 + i * 16 + rb + r) * DMODEL + col;
        X[idx] = X[idx] + acc[i][j][r];
      }
    }
}

// ---------------------------------------------------------------------------
// Head: rmsnorm(last token) -> elu -> @ out_W^T (256 -> 12)
// ---------------------------------------------------------------------------
__global__ void k_head(const float* __restrict__ X, const float* __restrict__ nfw,
                       const float* __restrict__ outW, float* __restrict__ out) {
  __shared__ float sy[DMODEL];
  __shared__ float wsum[8];
  int b = blockIdx.x, t = threadIdx.x;
  size_t row = (size_t)b * 512 + 511;
  float x = X[row * DMODEL + t];
  float s2 = wave_sum(x * x);
  int wv = t >> 5, lane = t & 31;
  if (lane == 0) wsum[wv] = s2;
  __syncthreads();
  float tot = 0.f;
#pragma unroll
  for (int i = 0; i < 8; i++) tot += wsum[i];
  float y = x * rsqrtf(tot * (1.f / DMODEL) + 1e-5f) * nfw[t];
  y = (y > 0.f) ? y : (expf(y) - 1.f);
  sy[t] = y;
  __syncthreads();
  if (t < 12) {
    float a = 0.f;
    for (int k = 0; k < DMODEL; k++) a += sy[k] * outW[t * DMODEL + k];
    out[b * 12 + t] = a;
  }
}

// ---------------------------------------------------------------------------
// Host orchestration
// ---------------------------------------------------------------------------
extern "C" void kernel_launch(void* const* d_in, const int* in_sizes, int n_in,
                              void* d_out, int out_size, void* d_ws, size_t ws_size,
                              hipStream_t stream) {
  (void)in_sizes; (void)n_in; (void)out_size; (void)ws_size;
  const float* obs  = (const float*)d_in[0];
  const float* inW  = (const float*)d_in[1];
  const float* ln1w = (const float*)d_in[2];
  const float* ln1b = (const float*)d_in[3];
  const float* rmsw = (const float*)d_in[4];   // (4,256)
  const float* WiF  = (const float*)d_in[5];   // (4,1160,256)
  const float* cwF  = (const float*)d_in[6];   // (4,640,4)
  const float* cbF  = (const float*)d_in[7];   // (4,640)
  const float* dtbF = (const float*)d_in[8];   // (4,8)
  const float* AlF  = (const float*)d_in[9];   // (4,8)
  const float* DpF  = (const float*)d_in[10];  // (4,8)
  const float* gwF  = (const float*)d_in[11];  // (4,512)
  const float* WoF  = (const float*)d_in[12];  // (4,256,512)
  const float* nfw  = (const float*)d_in[13];
  const float* outW = (const float*)d_in[14];
  float* out = (float*)d_out;

  char* w = (char*)d_ws;
  size_t off = 0;
  auto alloc = [&](size_t bytes) -> char* {
    char* p = w + off;
    off += (bytes + 255) & ~(size_t)255;
    return p;
  };
  float*  X    = (float*) alloc((size_t)NROWS * DMODEL * 4);
  __bf16* Hbf  = (__bf16*)alloc((size_t)NROWS * DMODEL * 2);
  __bf16* WiB  = (__bf16*)alloc((size_t)4 * NPROJP * DMODEL * 2);
  __bf16* WoB  = (__bf16*)alloc((size_t)4 * DMODEL * DINNER * 2);
  float*  ZX   = (float*) alloc((size_t)NROWS * NPROJ * 4);
  __bf16* XDT  = (__bf16*)alloc((size_t)NROWS * DINNER * 2);
  float*  XS   = (float*) alloc((size_t)NROWS * DINNER * 4);
  __bf16* Bb   = (__bf16*)alloc((size_t)NROWS * NSTATE * 2);
  __bf16* Cbb  = (__bf16*)alloc((size_t)NROWS * NSTATE * 2);
  float*  ADT  = (float*) alloc((size_t)NROWS * NHEADS * 4);
  float*  ACUM = (float*) alloc((size_t)NBATCH * NHEADS * NCHUNK * CHUNK * 4);
  float*  CSUM = (float*) alloc((size_t)NBATCH * NHEADS * NCHUNK * 4);
  float*  SBUF = (float*) alloc((size_t)NBATCH * NHEADS * NCHUNK * HD * NSTATE * 4);
  __bf16* PBUF = (__bf16*)alloc((size_t)NBATCH * NHEADS * NCHUNK * HD * NSTATE * 2);
  float*  YD   = (float*) alloc((size_t)NROWS * DINNER * 4);
  __bf16* YG   = (__bf16*)alloc((size_t)NROWS * DINNER * 2);

  k_wcvt<<<dim3(4 * NPROJP), 256, 0, stream>>>(WiF, WoF, WiB, WoB);
  k_embed_ln<<<dim3(NROWS), 256, 0, stream>>>(obs, inW, ln1w, ln1b, X);

  for (int L = 0; L < 4; L++) {
    k_rms<<<dim3(NROWS / 8), 256, 0, stream>>>(X, rmsw + L * DMODEL, Hbf);
    k_gemm_in<<<dim3(NROWS / 64, NPROJP / 256), 256, 0, stream>>>(
        Hbf, WiB + (size_t)L * NPROJP * DMODEL, ZX);
    k_conv<<<dim3(NROWS), 640, 0, stream>>>(
        ZX, cwF + (size_t)L * 640 * 4, cbF + L * 640, dtbF + L * NHEADS,
        AlF + L * NHEADS, XDT, XS, Bb, Cbb, ADT);
    k_cumsum<<<dim3(NBATCH * NHEADS * NCHUNK), CHUNK, 0, stream>>>(ADT, ACUM, CSUM);
    k_chunk<<<dim3(NBATCH * NCHUNK * NHEADS), 128, 0, stream>>>(
        Cbb, Bb, XDT, ACUM, YD, SBUF);
    k_scan<<<dim3(NBATCH * NHEADS), 256, 0, stream>>>(SBUF, CSUM, PBUF);
    k_yoff<<<dim3(NBATCH * NCHUNK * NHEADS), 128, 0, stream>>>(
        Cbb, PBUF, ACUM, XS, DpF + L * NHEADS, YD);
    k_gate<<<dim3(NROWS / 8), 256, 0, stream>>>(YD, ZX, gwF + L * DINNER, YG);
    k_gemm_out<<<dim3(NROWS / 64, 1), 256, 0, stream>>>(
        YG, WoB + (size_t)L * DMODEL * DINNER, X);
  }

  k_head<<<dim3(NBATCH), 256, 0, stream>>>(X, nfw, outW, out);
}
